// LearnableWatershedWithSDF_8315056685583
// MI455X (gfx1250) — matrix operand order repaired and checked by reading, not compiled, not run on hardware
//
#include <hip/hip_runtime.h>
#include <hip/hip_bf16.h>

#define HH 512
#define WW 512
#define BB 2
#define HW (HH*WW)          // 262144 = 1<<18
#define NPIX (BB*HW)        // 524288
#define NM 16

typedef __attribute__((ext_vector_type(16))) _Float16 v16h;
typedef __attribute__((ext_vector_type(8)))  _Float16 v8h;
typedef __attribute__((ext_vector_type(8)))  float    v8f;

union V16H { v16h v; v8h h[2]; _Float16 f[16]; };
union V8F  { v8f  v; float f[8]; };

// ---------------------------------------------------------------- grayscale
__global__ void k_gray(const float* __restrict__ img, float* __restrict__ gray) {
    int p = blockIdx.x * blockDim.x + threadIdx.x;
    if (p >= NPIX) return;
    int b  = p >> 18;
    int yx = p & (HW - 1);
    const float* base = img + (size_t)b * 3 * HW + yx;
    gray[p] = 0.2989f * base[0] + 0.587f * base[HW] + 0.114f * base[2 * HW];
}

// ------------------------------------------- weight repack: f32 -> f16 K-major
// W2T: [32 cout][160 kpad]  k = tap*16 + cin   (K=144 padded to 160)
// W3T: [16 cout][288 k]     k = tap*32 + cin
__global__ void k_prep_w(const float* __restrict__ w2, const float* __restrict__ w3,
                         _Float16* __restrict__ W2T, _Float16* __restrict__ W3T) {
    int i = blockIdx.x * blockDim.x + threadIdx.x;
    for (int j = i; j < 32 * 160; j += gridDim.x * blockDim.x) {
        int cout = j / 160, k = j % 160;
        float v = 0.f;
        if (k < 144) { int tap = k >> 4, cin = k & 15; v = w2[(cout * 16 + cin) * 9 + tap]; }
        W2T[j] = (_Float16)v;
    }
    for (int j = i; j < 16 * 288; j += gridDim.x * blockDim.x) {
        int cout = j / 288, k = j % 288;
        int tap = k >> 5, cin = k & 31;
        W3T[j] = (_Float16)w3[(cout * 32 + cin) * 9 + tap];
    }
}

// ---------------------------------------------------------------- conv1 (K=9)
__global__ void k_conv1(const float* __restrict__ gray,
                        const float* __restrict__ w1, const float* __restrict__ b1,
                        _Float16* __restrict__ x1) {
    int p = blockIdx.x * blockDim.x + threadIdx.x;
    if (p >= NPIX) return;
    int x = p & (WW - 1);
    int y = (p >> 9) & (HH - 1);
    int b = p >> 18;
    float g[9];
    #pragma unroll
    for (int t = 0; t < 9; t++) {
        int ny = y + t / 3 - 1, nx = x + t % 3 - 1;
        g[t] = ((unsigned)ny < HH && (unsigned)nx < WW)
             ? gray[(b << 18) + (ny << 9) + nx] : 0.f;
    }
    V16H o;
    #pragma unroll
    for (int oc = 0; oc < 16; oc++) {
        float acc = b1[oc];
        #pragma unroll
        for (int t = 0; t < 9; t++) acc += w1[oc * 9 + t] * g[t];
        o.f[oc] = (_Float16)fmaxf(acc, 0.f);
    }
    *(v16h*)(x1 + (size_t)p * 16) = o.v;
}

// 8 contiguous-half load with zero padding (NHWC, 16ch records)
__device__ __forceinline__ v8h ldx1(const _Float16* __restrict__ x1,
                                    int b, int ny, int nx, int halfOff) {
    if ((unsigned)ny < HH && (unsigned)nx < WW)
        return *(const v8h*)(x1 + ((size_t)((b << 18) + (ny << 9) + nx) << 4) + halfOff);
    v8h z = {};
    return z;
}

// ------------------------------------- conv2: implicit GEMM via WMMA f16->f32
// per wave: 16 pixels x 32 couts; K = 144 (5 chunks of 32, chunk = 2 taps)
__global__ void __launch_bounds__(256) k_conv2(const _Float16* __restrict__ x1,
                                               const _Float16* __restrict__ W2T,
                                               const float* __restrict__ b2,
                                               _Float16* __restrict__ x2) {
    __shared__ float tile[8][16][32];
    int lane  = threadIdx.x & 31;
    int wave  = threadIdx.x >> 5;
    int strip = blockIdx.x * 8 + wave;       // 32768 strips of 16 pixels
    int p0 = strip << 4;
    int xb = p0 & (WW - 1);
    int y  = (p0 >> 9) & (HH - 1);
    int b  = p0 >> 18;
    int m  = lane & 15;                      // pixel within strip / cout within tile
    int h  = lane >> 4;                      // lane-half selects K sub-range
    int x  = xb + m;

    // warm next strip's activation row in GL2 (global_prefetch_b8)
    __builtin_prefetch(x1 + ((size_t)(p0 + 128) << 4), 0, 1);

    float bv0 = b2[m], bv1 = b2[16 + m];
    v8f c0 = {bv0, bv0, bv0, bv0, bv0, bv0, bv0, bv0};
    v8f c1 = {bv1, bv1, bv1, bv1, bv1, bv1, bv1, bv1};

    #pragma unroll
    for (int ch = 0; ch < 5; ch++) {
        int t0 = 2 * ch, t1 = 2 * ch + 1;
        V16H a;
        a.h[0] = ldx1(x1, b, y + t0 / 3 - 1, x + t0 % 3 - 1, h * 8);   // K = ch*32 + {h*8..}
        if (t1 < 9)
            a.h[1] = ldx1(x1, b, y + t1 / 3 - 1, x + t1 % 3 - 1, h * 8); // K = ch*32+16+{h*8..}
        else { v8h z = {}; a.h[1] = z; }
        int kb = ch * 32 + h * 8;            // B: K = kb..kb+7 and kb+16..kb+23 (matches A)
        V16H wb0u, wb1u;
        wb0u.h[0] = *(const v8h*)(W2T + (size_t)m * 160 + kb);
        wb0u.h[1] = *(const v8h*)(W2T + (size_t)m * 160 + kb + 16);
        wb1u.h[0] = *(const v8h*)(W2T + (size_t)(16 + m) * 160 + kb);
        wb1u.h[1] = *(const v8h*)(W2T + (size_t)(16 + m) * 160 + kb + 16);
        v16h wb0 = wb0u.v, wb1 = wb1u.v;
        c0 = __builtin_amdgcn_wmma_f32_16x16x32_f16(false, a.v, false, wb0, (short)0, c0, false, false);
        c1 = __builtin_amdgcn_wmma_f32_16x16x32_f16(false, a.v, false, wb1, (short)0, c1, false, false);
    }
    V8F d0, d1; d0.v = c0; d1.v = c1;
    int mrow = (lane < 16) ? 0 : 8;
    #pragma unroll
    for (int r = 0; r < 8; r++) {
        tile[wave][r + mrow][m]      = d0.f[r];
        tile[wave][r + mrow][16 + m] = d1.f[r];
    }
    __syncthreads();
    int pp = threadIdx.x >> 1;               // pixel within block (0..127)
    int hh = threadIdx.x & 1;                // cout half
    V16H o;
    #pragma unroll
    for (int j = 0; j < 16; j++)
        o.f[j] = (_Float16)fmaxf(tile[pp >> 4][pp & 15][hh * 16 + j], 0.f);
    *(v16h*)(x2 + ((size_t)(blockIdx.x * 128 + pp) << 5) + hh * 16) = o.v;
}

// --------------------- conv3 + argmax -> markers; K = 288 (9 chunks, 1 tap each)
__global__ void __launch_bounds__(256) k_conv3(const _Float16* __restrict__ x2,
                                               const _Float16* __restrict__ W3T,
                                               const float* __restrict__ b3,
                                               int* __restrict__ markers) {
    __shared__ float tile[8][16][16];
    int lane  = threadIdx.x & 31;
    int wave  = threadIdx.x >> 5;
    int strip = blockIdx.x * 8 + wave;
    int p0 = strip << 4;
    int xb = p0 & (WW - 1);
    int y  = (p0 >> 9) & (HH - 1);
    int b  = p0 >> 18;
    int m  = lane & 15;
    int h  = lane >> 4;
    int x  = xb + m;

    float bv = b3[m];
    v8f c = {bv, bv, bv, bv, bv, bv, bv, bv};
    #pragma unroll
    for (int t = 0; t < 9; t++) {
        int ny = y + t / 3 - 1, nx = x + t % 3 - 1;
        V16H a;
        if ((unsigned)ny < HH && (unsigned)nx < WW) {
            const _Float16* rec = x2 + ((size_t)((b << 18) + (ny << 9) + nx) << 5);
            a.h[0] = *(const v8h*)(rec + h * 8);        // cin  h*8..h*8+7   (K=h*8..)
            a.h[1] = *(const v8h*)(rec + 16 + h * 8);   // cin 16+h*8..      (K=16+h*8..)
        } else { v8h z = {}; a.h[0] = z; a.h[1] = z; }
        V16H wbu;
        wbu.h[0] = *(const v8h*)(W3T + (size_t)m * 288 + t * 32 + h * 8);
        wbu.h[1] = *(const v8h*)(W3T + (size_t)m * 288 + t * 32 + 16 + h * 8);
        v16h wb = wbu.v;
        c = __builtin_amdgcn_wmma_f32_16x16x32_f16(false, a.v, false, wb, (short)0, c, false, false);
    }
    V8F d; d.v = c;
    int mrow = (lane < 16) ? 0 : 8;
    #pragma unroll
    for (int r = 0; r < 8; r++) tile[wave][r + mrow][m] = d.f[r];
    __syncthreads();
    int t2 = threadIdx.x;
    if (t2 < 128) {
        int w = t2 >> 4, pm = t2 & 15;
        float best = tile[w][pm][0]; int bi = 0;
        #pragma unroll
        for (int j = 1; j < 16; j++) {
            float v = tile[w][pm][j];
            if (v > best) { best = v; bi = j; }          // jnp.argmax: first max wins
        }
        markers[blockIdx.x * 128 + t2] = bi;
    }
}

// ---------------------------------------------------------------- one-hot init
__global__ void k_onehot(const int* __restrict__ markers, _Float16* __restrict__ lab) {
    int p = blockIdx.x * blockDim.x + threadIdx.x;
    if (p >= NPIX) return;
    int mk = markers[p];
    V16H o;
    #pragma unroll
    for (int j = 0; j < 16; j++) o.f[j] = (_Float16)(j == mk ? 1.f : 0.f);
    *(v16h*)(lab + (size_t)p * 16) = o.v;
}

// ------------- flood: softmax(avg_pool3(lab)) — bias is softmax-shift-invariant
// 16x16-pixel tile per block; 18x18-record halo staged into LDS with
// GLOBAL_LOAD_ASYNC_TO_LDS_B128 (ASYNCcnt), consumed via ds_load_b128.
__global__ void __launch_bounds__(256) k_flood(const _Float16* __restrict__ in,
                                               _Float16* __restrict__ out) {
    __shared__ __align__(64) _Float16 smem[18 * 18 * 16];   // 10.125 KB
    int tid = threadIdx.x;
    int bx  = blockIdx.x;                    // B*32*32 = 2048 blocks
    int b   = bx >> 10;
    int ty  = (bx >> 5) & 31;
    int tx  = bx & 31;
    int y0  = ty << 4, x0 = tx << 4;

    unsigned smemBase = (unsigned)(size_t)(&smem[0]);  // LDS aperture: low 32 bits = LDS offset
    for (int idx = tid; idx < 324; idx += 256) {
        int ly = idx / 18, lx = idx - ly * 18;
        int gy = y0 - 1 + ly, gx = x0 - 1 + lx;
        unsigned ldsOff = smemBase + (unsigned)idx * 32u;
        if ((unsigned)gy < HH && (unsigned)gx < WW) {
            const _Float16* g = in + ((size_t)((b << 18) + (gy << 9) + gx) << 4);
            asm volatile("global_load_async_to_lds_b128 %0, %1, off"
                         :: "v"(ldsOff), "v"(g) : "memory");
        } else {
            v16h z = {};
            *(v16h*)(smem + idx * 16) = z;   // zero padding (DScnt path)
        }
    }
    asm volatile("s_wait_asynccnt 0x0" ::: "memory");
    __syncthreads();

    int ly = tid >> 4, lx = tid & 15;
    float acc[16];
    #pragma unroll
    for (int j = 0; j < 16; j++) acc[j] = 0.f;
    #pragma unroll
    for (int dy = 0; dy < 3; dy++) {
        #pragma unroll
        for (int dx = 0; dx < 3; dx++) {
            V16H v;
            v.v = *(const v16h*)(smem + ((ly + dy) * 18 + (lx + dx)) * 16);
            #pragma unroll
            for (int j = 0; j < 16; j++) acc[j] += (float)v.f[j];
        }
    }
    float mx = -1e30f;
    #pragma unroll
    for (int j = 0; j < 16; j++) { acc[j] *= (1.f / 9.f); mx = fmaxf(mx, acc[j]); }
    float e[16], s = 0.f;
    #pragma unroll
    for (int j = 0; j < 16; j++) { e[j] = __expf(acc[j] - mx); s += e[j]; }
    float r = __builtin_amdgcn_rcpf(s);
    V16H o;
    #pragma unroll
    for (int j = 0; j < 16; j++) o.f[j] = (_Float16)(e[j] * r);
    int p = (b << 18) + ((y0 + ly) << 9) + (x0 + lx);
    *(v16h*)(out + (size_t)p * 16) = o.v;
}

// ---------------------------------------------------------------- final argmax
__global__ void k_out(const _Float16* __restrict__ lab, float* __restrict__ outp) {
    int p = blockIdx.x * blockDim.x + threadIdx.x;
    if (p >= NPIX) return;
    V16H v; v.v = *(const v16h*)(lab + (size_t)p * 16);
    float best = (float)v.f[0]; int bi = 0;
    #pragma unroll
    for (int j = 1; j < 16; j++) {
        float c = (float)v.f[j];
        if (c > best) { best = c; bi = j; }
    }
    outp[p] = (float)bi;
}

extern "C" void kernel_launch(void* const* d_in, const int* in_sizes, int n_in,
                              void* d_out, int out_size, void* d_ws, size_t ws_size,
                              hipStream_t stream) {
    const float* img = (const float*)d_in[0];
    // d_in[1]=Kx, d_in[2]=Ky: dead code (softmax shift invariance), unused
    const float* w1 = (const float*)d_in[3];
    const float* b1 = (const float*)d_in[4];
    const float* w2 = (const float*)d_in[5];
    const float* b2 = (const float*)d_in[6];
    const float* w3 = (const float*)d_in[7];
    const float* b3 = (const float*)d_in[8];

    char* ws = (char*)d_ws;
    size_t off = 0;
    auto carve = [&](size_t bytes) -> void* {
        void* p = ws + off;
        off = (off + bytes + 255) & ~(size_t)255;
        return p;
    };
    float*    gray = (float*)carve((size_t)NPIX * 4);          // region A (dead after conv1)
    _Float16* x1   = (_Float16*)carve((size_t)NPIX * 16 * 2);  // region A (dead after conv2)
    _Float16* x2   = (_Float16*)carve((size_t)NPIX * 32 * 2);  // region B (dead after conv3)
    int*      mk   = (int*)carve((size_t)NPIX * 4);
    _Float16* W2T  = (_Float16*)carve(32 * 160 * 2);
    _Float16* W3T  = (_Float16*)carve(16 * 288 * 2);
    // alias label ping-pong over dead conv activations (saves ~34 MB)
    _Float16* lab0 = (_Float16*)gray;   // needs 16.8 MB; gray+x1 region = 18.8 MB
    _Float16* lab1 = (_Float16*)x2;    // needs 16.8 MB; x2 region = 33.6 MB

    dim3 blk(256);
    dim3 gN((NPIX + 255) / 256);     // 2048
    dim3 gW(NPIX / 128);             // 4096 (8 waves x 16-pixel strips per block)
    dim3 gT(BB * 32 * 32);           // 2048 (16x16-pixel tiles)

    k_gray  <<<gN, blk, 0, stream>>>(img, gray);
    k_prep_w<<<dim3(20), blk, 0, stream>>>(w2, w3, W2T, W3T);
    k_conv1 <<<gN, blk, 0, stream>>>(gray, w1, b1, x1);
    k_conv2 <<<gW, blk, 0, stream>>>(x1, W2T, b2, x2);
    k_conv3 <<<gW, blk, 0, stream>>>(x2, W3T, b3, mk);
    k_onehot<<<gN, blk, 0, stream>>>(mk, lab0);

    const _Float16* fin = lab0;
    _Float16* fout = lab1;
    for (int i = 0; i < 50; i++) {
        k_flood<<<gT, blk, 0, stream>>>(fin, fout);
        const _Float16* t = fout;
        fout = (_Float16*)fin;
        fin = t;
    }
    k_out<<<gN, blk, 0, stream>>>(fin, (float*)d_out);
}
